// QGNNLayer_v2_20555713479334
// MI455X (gfx1250) — matrix operand order repaired and checked by reading, not compiled, not run on hardware
//
#include <hip/hip_runtime.h>
#include <hip/hip_bf16.h>
#include <math.h>
#include <stdint.h>

typedef unsigned short u16;
typedef unsigned int   u32;
typedef __attribute__((ext_vector_type(16))) __bf16 v16bf;
typedef __attribute__((ext_vector_type(8)))  float  v8f;

#define N_NODES 50000
#define KDIM    1024
#define ODIM    256
#define NEDGE   800000
#define BN_EPS  1e-5f

// workspace layout (bytes)
#define HAM_OFF   0                                    // u16[ODIM*KDIM] hamilton^T bf16 (512 KB)
#define AGG_OFF   ((size_t)ODIM * KDIM * 2)            // float[N_NODES*ODIM] (51.2 MB)
#define SUMS_OFF  (AGG_OFF + (size_t)N_NODES * ODIM * 4)
#define SUMSQ_OFF (SUMS_OFF + (size_t)ODIM * 4)

// 32-dim hypercomplex multiplication table (from the reference)
__constant__ int KERN[32][32] = {
{1,2,3,4,5,6,7,8,9,10,11,12,13,14,15,16,17,18,19,20,21,22,23,24,25,26,27,28,29,30,31,32},
{2,-1,4,-3,6,-5,-8,7,10,-9,-12,11,-14,13,16,-15,18,-17,-20,19,-22,21,24,-23,-26,25,28,-27,30,-29,-32,31},
{3,-4,-1,2,7,8,-5,-6,11,12,-9,-10,-15,-16,13,14,19,20,-17,-18,-23,-24,21,22,-27,-28,25,26,31,32,-29,-30},
{4,3,-2,-1,8,-7,6,-5,12,-11,10,-9,-16,15,-14,13,20,-19,18,-17,-24,23,-22,21,-28,27,-26,25,32,-31,30,-29},
{5,-6,-7,-8,-1,2,3,4,13,14,15,16,-9,-10,-11,-12,21,22,23,24,-17,-18,-19,-20,-29,-30,-31,-32,25,26,27,28},
{6,5,-8,7,-2,-1,-4,3,14,-13,16,-15,10,-9,12,-11,22,-21,24,-23,18,-17,20,-19,-30,29,-32,31,-26,25,-28,27},
{7,8,5,-6,-3,4,-1,-2,15,-16,-13,14,11,-12,-9,10,23,-24,-21,22,19,-20,-17,18,-31,32,29,-30,-27,28,25,-26},
{8,-7,6,5,-4,-3,2,-1,16,15,-14,-13,12,11,-10,-9,24,23,-22,-21,20,19,-18,-17,-32,-31,30,29,-28,-27,26,25},
{9,-10,-11,-12,-13,-14,-15,-16,-1,2,3,4,5,6,7,8,25,26,27,28,29,30,31,32,-17,-18,-19,-20,-21,-22,-23,-24},
{10,9,-12,11,-14,13,16,-15,-2,-1,-4,3,-6,5,8,-7,26,-25,28,-27,30,-29,-32,31,18,-17,20,-19,22,-21,-24,23},
{11,12,9,-10,-15,-16,13,14,-3,4,-1,-2,-7,-8,5,6,27,-28,-25,26,31,32,-29,-30,19,-20,-17,18,23,24,-21,-22},
{12,-11,10,9,-16,15,-14,13,-4,-3,2,-1,-8,7,-6,5,28,27,-26,-25,32,-31,30,-29,20,19,-18,-17,24,-23,22,-21},
{13,14,15,16,9,-10,-11,-12,-5,6,7,8,-1,-2,-3,-4,29,-30,-31,-32,-25,26,27,28,21,-22,-23,-24,-17,18,19,20},
{14,-13,16,-15,10,9,12,-11,-6,-5,8,-7,2,-1,4,-3,30,29,-32,31,-26,-25,-28,27,22,21,-24,23,-18,-17,-20,19},
{15,-16,-13,14,11,-12,9,10,-7,-8,-5,6,3,-4,-1,2,31,32,29,-30,-27,28,-25,-26,23,24,21,-22,-19,20,-17,-18},
{16,15,-14,-13,12,11,-10,9,-8,7,-6,-5,4,3,-2,-1,32,-31,30,29,-28,-27,26,-25,24,-23,22,21,-20,-19,18,-17},
{17,-18,-19,-20,-21,-22,-23,-24,-25,-26,-27,-28,-29,-30,-31,-32,-1,2,3,4,5,6,7,8,9,10,11,12,13,14,15,16},
{18,17,-20,19,-22,21,24,-23,-26,25,28,-27,30,-29,-32,31,-2,-1,-4,3,-6,5,8,-7,-10,9,12,-11,14,-13,-16,15},
{19,20,17,-18,-23,-24,21,22,-27,-28,25,26,31,32,-29,-30,-3,4,-1,-2,-7,-8,5,6,-11,-12,9,10,15,16,-13,-14},
{20,-19,18,17,-24,23,-22,21,-28,27,-26,25,32,-31,30,-29,-4,-3,2,-1,-8,7,-6,5,-12,11,-10,9,16,-15,14,-13},
{21,22,23,24,17,-18,-19,-20,-29,-30,-31,-32,25,26,27,28,-5,6,7,8,-1,-2,-3,-4,-13,-14,-15,-16,9,10,11,12},
{22,-21,24,-23,18,17,20,-19,-30,29,-32,31,-26,25,-28,27,-6,-5,8,-7,2,-1,4,-3,-14,13,-16,15,-10,9,-12,11},
{23,-24,-21,22,19,-20,17,18,-31,32,29,-30,-27,28,25,-26,-7,-8,-5,6,3,-4,-1,2,-15,16,13,-14,-11,12,9,-10},
{24,23,-22,-21,20,19,-18,17,-32,-31,30,29,-28,-27,26,25,-8,7,-6,-5,4,3,-2,-1,-16,-15,14,13,-12,-11,10,9},
{25,26,27,28,29,30,31,32,17,-18,-19,-20,-21,-22,-23,-24,-9,10,11,12,13,14,15,16,-1,-2,-3,-4,-5,-6,-7,-8},
{26,-25,28,-27,30,-29,-32,31,18,17,20,-19,22,-21,-24,23,-10,-9,12,-11,14,-13,-16,15,2,-1,4,-3,6,-5,-8,7},
{27,-28,-25,26,31,32,-29,-30,19,-20,17,18,23,24,-21,-22,-11,-12,-9,10,15,16,-13,-14,3,-4,-1,2,7,8,-5,-6},
{28,27,-26,-25,32,-31,30,-29,20,19,-18,17,24,-23,22,-21,-12,11,-10,-9,16,-15,14,-13,4,3,-2,-1,8,-7,6,-5},
{29,-30,-31,-32,-25,26,27,28,21,-22,-23,-24,17,18,19,20,-13,-14,-15,-16,-9,10,11,12,5,-6,-7,-8,-1,2,3,4},
{30,29,-32,31,-26,-25,-28,27,22,21,-24,23,-18,17,-20,19,-14,13,-16,15,-10,-9,-12,11,6,5,-8,7,-2,-1,-4,3},
{31,32,29,-30,-27,28,-25,-26,23,24,21,-22,-19,20,17,-18,-15,16,13,-14,-11,12,-9,-10,7,8,5,-6,-3,4,-1,-2},
{32,-31,30,29,-28,-27,26,-25,24,-23,22,21,-20,-19,18,17,-16,-15,14,13,-12,-11,10,-9,8,-7,6,5,-4,-3,2,-1}
};

__device__ __forceinline__ u16 f2bf(float f) {  // round-to-nearest-even fp32 -> bf16 bits
  u32 u = __float_as_uint(f);
  u32 r = u + 0x7FFFu + ((u >> 16) & 1u);
  return (u16)(r >> 16);
}

// LDS byte offset of a generic pointer to __shared__ (ISA 10.2: the low 32 bits of the address)
__device__ __forceinline__ u32 lds_off(const void* p) {
  return (u32)(uintptr_t)p;
}

// ---------------- zero workspace accumulators ----------------
__global__ __launch_bounds__(256) void k_zero(float* __restrict__ p, size_t n) {
  size_t i = (size_t)blockIdx.x * 256 + threadIdx.x;
  if (i < n) p[i] = 0.0f;
}

// ---------------- build hamilton^T (256 x 1024) in bf16 ----------------
// hamilton[a*32+i, b*8+d] = sign(K[a][b]) * weight[i, (|K[a][b]|-1)*8 + d]
__global__ __launch_bounds__(256) void k_hamilton(const float* __restrict__ w, u16* __restrict__ HT) {
  int flat = blockIdx.x * 256 + threadIdx.x;   // flat = c*1024 + r, 0..262143
  int r = flat & (KDIM - 1);
  int c = flat >> 10;
  int a = r >> 5, i = r & 31;
  int b = c >> 3, d = c & 7;
  int kv = KERN[a][b];
  int idx = (kv < 0 ? -kv : kv) - 1;
  float wv = w[i * ODIM + idx * 8 + d];
  HT[flat] = f2bf(kv < 0 ? -wv : wv);
}

// ---------------- GEMM: support = input(50000x1024,f32) @ hamilton(1024x256,bf16) ----------------
// Block tile 128(M) x 128(N), 8 waves; wave = 32x64 patch = 2x4 accumulators of 16x16.
// Double-buffered LDS; B tile staged with GLOBAL_LOAD_ASYNC_TO_LDS_B128 (ASYNCcnt),
// A tile converted fp32->bf16 in-register, prefetched 2 k-steps ahead.
__global__ __launch_bounds__(256) void k_gemm_wmma(const float* __restrict__ A,
                                                   const u16* __restrict__ HT,
                                                   float* __restrict__ C) {
  __shared__ u16 As[2][128][40];   // M x K tiles, bf16, +8 pad (rows stay 16B aligned)
  __shared__ u16 Bs[2][128][40];   // N x K tiles, bf16

  const int tid   = threadIdx.x;
  const int lane  = tid & 31;
  const int wave  = tid >> 5;
  const int waveM = wave & 3;          // 4 waves along M
  const int waveN = wave >> 2;         // 2 waves along N
  const int mBase = blockIdx.x * 128;
  const int nBase = blockIdx.y * 128;

  v8f acc[2][4] = {};

  const int lr = tid >> 1;             // tile row (A) / tile col (B): 0..127
  const int lh = (tid & 1) * 16;       // K half: 0 or 16
  const int grow = mBase + lr;
  const bool arow_ok = (grow < N_NODES);
  const float* Abase = A + (size_t)grow * KDIM + lh;
  const u16*   Bbase = HT + (size_t)(nBase + lr) * KDIM + lh;

  // ---- staging helpers ----
  auto stageA = [&](int k0, int buf) {
    alignas(16) u16 tmp[16];
    if (arow_ok) {
      const float4* src = (const float4*)(Abase + k0);
      if (k0 + 64 < KDIM) __builtin_prefetch(Abase + k0 + 64, 0, 0);   // global_prefetch_b8
      #pragma unroll
      for (int q = 0; q < 4; ++q) {
        float4 f = src[q];
        tmp[q*4+0] = f2bf(f.x); tmp[q*4+1] = f2bf(f.y);
        tmp[q*4+2] = f2bf(f.z); tmp[q*4+3] = f2bf(f.w);
      }
    } else {
      #pragma unroll
      for (int q = 0; q < 16; ++q) tmp[q] = 0;
    }
    *(uint4*)&As[buf][lr][lh]     = *(const uint4*)&tmp[0];
    *(uint4*)&As[buf][lr][lh + 8] = *(const uint4*)&tmp[8];
  };
  auto stageB = [&](int k0, int buf) {
    // async copy 32 bytes (2 x b128) straight into LDS, no VGPR round trip
    const u16* g0 = Bbase + k0;
    u32 l0 = lds_off(&Bs[buf][lr][lh]);
    asm volatile("global_load_async_to_lds_b128 %0, %1, off"
                 :: "v"(l0), "v"(g0) : "memory");
    const u16* g1 = g0 + 8;
    u32 l1 = l0 + 16;
    asm volatile("global_load_async_to_lds_b128 %0, %1, off"
                 :: "v"(l1), "v"(g1) : "memory");
  };

  // prologue: stage first tile into buffer 0
  stageA(0, 0);
  stageB(0, 0);
  asm volatile("s_wait_asynccnt 0x0" ::: "memory");
  __syncthreads();

  const int kbA = (lane < 16) ? 0 : 8;    // A frag: lanes 0-15 -> K 0..7 & 16..23; 16-31 -> 8..15 & 24..31
  const int kbB = (lane < 16) ? 0 : 8;    // B frag: lanes 0-15 -> K 0..7 & 16..23; 16-31 -> 8..15 & 24..31
  const int ml  = lane & 15;

  int buf = 0;
  for (int k0 = 0; k0 < KDIM; k0 += 32) {
    // issue next tile's loads early (overlap with WMMA burst below)
    const int nk = k0 + 32;
    if (nk < KDIM) {
      stageA(nk, buf ^ 1);
      stageB(nk, buf ^ 1);
    }

    // ---- compute on current buffer ----
    union Frag { v16bf v; uint4 q[2]; };
    Frag a[2];
    #pragma unroll
    for (int ms = 0; ms < 2; ++ms) {
      const int am = waveM * 32 + ms * 16 + ml;
      a[ms].q[0] = *(const uint4*)&As[buf][am][kbA];
      a[ms].q[1] = *(const uint4*)&As[buf][am][kbA + 16];
    }
    #pragma unroll
    for (int ns = 0; ns < 4; ++ns) {
      const int bn = waveN * 64 + ns * 16 + ml;
      Frag b;
      b.q[0] = *(const uint4*)&Bs[buf][bn][kbB];
      b.q[1] = *(const uint4*)&Bs[buf][bn][kbB + 16];
      #pragma unroll
      for (int ms = 0; ms < 2; ++ms) {
        acc[ms][ns] = __builtin_amdgcn_wmma_f32_16x16x32_bf16(
            false, a[ms].v, false, b.v, (short)0, acc[ms][ns], false, false);
      }
    }

    // next buffer fully written (asyncs + ds stores) before anyone reads it
    asm volatile("s_wait_asynccnt 0x0" ::: "memory");
    __syncthreads();
    buf ^= 1;
  }

  // store C (f32 16x16 C/D layout: N = lane%16; VGPR v -> M = v + (lane<16?0:8))
  const int mh = (lane < 16) ? 0 : 8;
  #pragma unroll
  for (int ms = 0; ms < 2; ++ms) {
    #pragma unroll
    for (int ns = 0; ns < 4; ++ns) {
      #pragma unroll
      for (int v = 0; v < 8; ++v) {
        int row = mBase + waveM * 32 + ms * 16 + mh + v;
        if (row < N_NODES)
          C[(size_t)row * ODIM + nBase + waveN * 64 + ns * 16 + ml] = acc[ms][ns][v];
      }
    }
  }
}

// ---------------- edge scatter: agg[row] += val * support[col] ----------------
#define EPB 64
__global__ __launch_bounds__(256) void k_scatter(const int* __restrict__ erow,
                                                 const int* __restrict__ ecol,
                                                 const float* __restrict__ ev,
                                                 const float* __restrict__ sup,
                                                 float* __restrict__ agg) {
  const int tid  = threadIdx.x;        // channel 0..255
  const int base = blockIdx.x * EPB;
  #pragma unroll 1
  for (int i = 0; i < EPB; ++i) {
    int e = base + i;
    if (e >= NEDGE) break;
    int r = erow[e];
    int c = ecol[e];
    float v = ev[e];
    float m = v * sup[(size_t)c * ODIM + tid];     // L2-resident gather
    unsafeAtomicAdd(&agg[(size_t)r * ODIM + tid], m);
  }
}

// ---------------- per-column sum / sumsq ----------------
#define RPB 64
__global__ __launch_bounds__(256) void k_stats(const float* __restrict__ agg,
                                               float* __restrict__ sums,
                                               float* __restrict__ sumsq) {
  const int c = threadIdx.x;
  int r0 = blockIdx.x * RPB;
  int r1 = r0 + RPB; if (r1 > N_NODES) r1 = N_NODES;
  float s = 0.0f, ss = 0.0f;
  for (int r = r0; r < r1; ++r) {
    float x = agg[(size_t)r * ODIM + c];
    s += x; ss += x * x;
  }
  unsafeAtomicAdd(&sums[c], s);
  unsafeAtomicAdd(&sumsq[c], ss);
}

// ---------------- batchnorm + tanh ----------------
__global__ __launch_bounds__(256) void k_bn_tanh(const float* __restrict__ agg,
                                                 const float* __restrict__ sums,
                                                 const float* __restrict__ sumsq,
                                                 const float* __restrict__ gamma,
                                                 const float* __restrict__ beta,
                                                 float* __restrict__ out) {
  size_t idx = (size_t)blockIdx.x * 256 + threadIdx.x;
  if (idx >= (size_t)N_NODES * ODIM) return;
  int c = (int)(idx & (ODIM - 1));
  const float inv_n = 1.0f / (float)N_NODES;
  float mean = sums[c] * inv_n;
  float var  = sumsq[c] * inv_n - mean * mean;
  float sc   = gamma[c] * rsqrtf(var + BN_EPS);
  float x    = (agg[idx] - mean) * sc + beta[c];
  out[idx] = tanhf(x);
}

extern "C" void kernel_launch(void* const* d_in, const int* in_sizes, int n_in,
                              void* d_out, int out_size, void* d_ws, size_t ws_size,
                              hipStream_t stream) {
  const float* input  = (const float*)d_in[0];
  const int*   erow   = (const int*)d_in[1];
  const int*   ecol   = (const int*)d_in[2];
  const float* ev     = (const float*)d_in[3];
  const float* weight = (const float*)d_in[4];
  const float* gamma  = (const float*)d_in[5];
  const float* beta   = (const float*)d_in[6];
  float* out = (float*)d_out;

  char* ws = (char*)d_ws;
  u16*   HT    = (u16*)  (ws + HAM_OFF);
  float* agg   = (float*)(ws + AGG_OFF);
  float* sums  = (float*)(ws + SUMS_OFF);
  float* sumsq = (float*)(ws + SUMSQ_OFF);

  // 1) zero accumulator + stats (contiguous region)
  size_t nz = (size_t)N_NODES * ODIM + 2 * ODIM;
  k_zero<<<(int)((nz + 255) / 256), 256, 0, stream>>>(agg, nz);

  // 2) build hamilton^T in bf16
  k_hamilton<<<(KDIM * ODIM) / 256, 256, 0, stream>>>(weight, HT);

  // 3) support = input @ hamilton  (bf16 WMMA, f32 accum) -> d_out
  dim3 g((N_NODES + 127) / 128, ODIM / 128);
  k_gemm_wmma<<<g, 256, 0, stream>>>(input, HT, out);

  // 4) edge gather/scale/scatter-add (L2 resident)
  k_scatter<<<(NEDGE + EPB - 1) / EPB, 256, 0, stream>>>(erow, ecol, ev, out, agg);

  // 5) column stats
  k_stats<<<(N_NODES + RPB - 1) / RPB, 256, 0, stream>>>(agg, sums, sumsq);

  // 6) normalize + tanh -> d_out
  size_t tot = (size_t)N_NODES * ODIM;
  k_bn_tanh<<<(int)((tot + 255) / 256), 256, 0, stream>>>(agg, sums, sumsq, gamma, beta, out);
}
